// AttentionPooling_75557064671340
// MI455X (gfx1250) — matrix operand order repaired and checked by reading, not compiled, not run on hardware
//
#include <hip/hip_runtime.h>
#include <hip/hip_bf16.h>
#include <math.h>

// ---------------------------------------------------------------------------
// AttentionPooling on MI455X (gfx1250, wave32, WMMA)
//   scores path: bf16 WMMA (v_wmma_f32_16x16x32_bf16), B-fragments preloaded
//                per nt-slab so the 8 WMMAs issue back-to-back (no per-op waits)
//   pooling path: f32, coalesced, memory-bound (~820 MB total -> ~35us @23.3TB/s)
// ---------------------------------------------------------------------------

typedef __attribute__((ext_vector_type(16))) __bf16 v16bf;
typedef __attribute__((ext_vector_type(8)))  __bf16 v8bf;
typedef __attribute__((ext_vector_type(8)))  float  v8f;

#define D_DIM 256
#define H_DIM 128
#define KT 8              // 256 / 32 k-steps
#define NT 8              // 128 / 16 n-tiles
#define ROWS_PER_BLOCK 128
#define LDS_STRIDE 264    // 256 + 8 bf16 pad: row stride 528B -> banks rotate by 4/row

#if __has_builtin(__builtin_amdgcn_tanhf)
__device__ __forceinline__ float fast_tanh(float x) { return __builtin_amdgcn_tanhf(x); }
#else
__device__ __forceinline__ float fast_tanh(float x) { return tanhf(x); }
#endif

__device__ __forceinline__ float fast_exp(float x) {
    // v_exp_f32 is exp2; scores - max <= 0 so no range reduction needed
    return __builtin_amdgcn_exp2f(x * 1.44269504088896341f);
}

// ---------------------------------------------------------------------------
// K0: pre-swizzle w1 (f32, [K=256][N=128]) into per-lane bf16 B-fragments:
//     wb[nt][kt][lane][i] = bf16( w1[kt*32 + (i&7) + 8*(lane>>4) + 16*(i>>3)][nt*16 + (lane&15)] )
// (B 32x16 bf16 layout: lanes 0-15 hold K {0-7,16-23}, lanes 16-31 hold K {8-15,24-31})
// nt-major so one nt-slab's 8 fragments are 8KB contiguous (single load clause).
// ---------------------------------------------------------------------------
__global__ void swizzle_w1(const float* __restrict__ w1, __bf16* __restrict__ wb) {
    int tid  = blockIdx.x * blockDim.x + threadIdx.x;   // 0..32767
    int i    = tid & 15;
    int lane = (tid >> 4) & 31;
    int kt   = (tid >> 9) & 7;
    int nt   = tid >> 12;
    int n = nt * 16 + (lane & 15);
    int k = kt * 32 + (i & 7) + 8 * (lane >> 4) + 16 * (i >> 3);
    wb[tid] = (__bf16)w1[k * H_DIM + n];
}

// ---------------------------------------------------------------------------
// K1: scores[i] = tanh(x[i]@w1 + b1) @ w2 + b2, via bf16 WMMA.
// Block = 256 threads = 8 waves, handles 128 rows. x tile staged in LDS as bf16.
// Each wave: A fragments (16x256) resident in 64 VGPRs; per nt-slab preload the
// 8 B fragments (64 VGPRs), then 8 chained WMMAs; fold tanh*w2 on the fly.
// ---------------------------------------------------------------------------
__global__ __launch_bounds__(256) void score_kernel(
    const float* __restrict__ x, const __bf16* __restrict__ wb,
    const float* __restrict__ b1, const float* __restrict__ w2,
    const float* __restrict__ b2, float* __restrict__ scores, int N)
{
    __shared__ __align__(16) __bf16 xs[ROWS_PER_BLOCK * LDS_STRIDE];
    const int t = threadIdx.x;
    const long long row0 = (long long)blockIdx.x * ROWS_PER_BLOCK;

    // ---- stage 128x256 f32 tile -> LDS bf16 (2 threads per row, b128 loads) ----
    {
        const int r  = t >> 1;
        const int ch = (t & 1) * 128;      // which 128-column half
        const long long rg = row0 + r;
        if (rg < N) {
            const float4* xr4 = (const float4*)(x + rg * D_DIM);
            #pragma unroll
            for (int c8 = 0; c8 < 16; ++c8) {
                float4 a = xr4[(ch >> 2) + c8 * 2];
                float4 b = xr4[(ch >> 2) + c8 * 2 + 1];
                v8bf p;
                p[0] = (__bf16)a.x; p[1] = (__bf16)a.y; p[2] = (__bf16)a.z; p[3] = (__bf16)a.w;
                p[4] = (__bf16)b.x; p[5] = (__bf16)b.y; p[6] = (__bf16)b.z; p[7] = (__bf16)b.w;
                *(v8bf*)&xs[r * LDS_STRIDE + ch + c8 * 8] = p;
            }
        } else {
            v8bf z = {};
            #pragma unroll
            for (int c8 = 0; c8 < 16; ++c8)
                *(v8bf*)&xs[r * LDS_STRIDE + ch + c8 * 8] = z;
        }
        // speculative prefetch of the next block's tile (global_prefetch_b8)
        if ((row0 + ROWS_PER_BLOCK) < N)
            __builtin_prefetch(x + (row0 + ROWS_PER_BLOCK) * D_DIM + t * 16, 0, 0);
    }
    __syncthreads();

    const int wave = t >> 5;
    const int lane = t & 31;
    const int half = lane >> 4;      // 0: lanes 0-15, 1: lanes 16-31
    const int nl   = lane & 15;
    // A-fragment source row for this lane (16-bit A layout: M = lane%16)
    const __bf16* xrow = &xs[(wave * 16 + nl) * LDS_STRIDE];

    // ---- load all A fragments into registers (8 x v16bf = 64 VGPRs) ----
    v16bf afr[KT];
    #pragma unroll
    for (int kt = 0; kt < KT; ++kt) {
        // A 16x32 bf16: lane half 0 covers K {0-7,16-23}, half 1 gets +8
        const int k0 = kt * 32 + 8 * half;
        v8bf lo = *(const v8bf*)&xrow[k0];
        v8bf hi = *(const v8bf*)&xrow[k0 + 16];
        afr[kt] = __builtin_shufflevector(lo, hi,
            0,1,2,3,4,5,6,7,8,9,10,11,12,13,14,15);
    }

    float acc[8];
    #pragma unroll
    for (int r = 0; r < 8; ++r) acc[r] = 0.f;
    const float b2v = b2[0];

    for (int nt = 0; nt < NT; ++nt) {
        // preload the full nt-slab of B fragments (8KB contiguous, L1-hot)
        const __bf16* wslab = &wb[(size_t)nt * KT * 32 * 16];
        v16bf bfr[KT];
        #pragma unroll
        for (int kt = 0; kt < KT; ++kt)
            bfr[kt] = *(const v16bf*)&wslab[(kt * 32 + lane) * 16];

        const float bias = b1[nt * 16 + nl];       // C layout: N = lane%16
        v8f c;
        #pragma unroll
        for (int r = 0; r < 8; ++r) c[r] = bias;

        #pragma unroll
        for (int kt = 0; kt < KT; ++kt)
            c = __builtin_amdgcn_wmma_f32_16x16x32_bf16(
                    false, afr[kt], false, bfr[kt], (short)0, c, false, false);

        // h = tanh(c); fold h @ w2 for this 16-column slab
        const float w2v = w2[nt * 16 + nl];
        #pragma unroll
        for (int r = 0; r < 8; ++r) acc[r] += fast_tanh(c[r]) * w2v;
    }

    // reduce over the 16 lanes of each half (the N dimension of the C tile)
    #pragma unroll
    for (int r = 0; r < 8; ++r) {
        float v = acc[r];
        v += __shfl_xor(v, 1, 32);
        v += __shfl_xor(v, 2, 32);
        v += __shfl_xor(v, 4, 32);
        v += __shfl_xor(v, 8, 32);
        acc[r] = v;
    }
    if (nl == 0) {
        // C layout: VGPR r holds row M = r + 8*half
        const long long rowbase = row0 + wave * 16 + 8 * half;
        #pragma unroll
        for (int r = 0; r < 8; ++r) {
            long long rr = rowbase + r;
            if (rr < N) scores[rr] = acc[r] + b2v;
        }
    }
}

// ---------------------------------------------------------------------------
// K2: segment boundaries via binary search in the sorted batch array.
// seg_start[g] = first index i with batch[i] >= g;  seg_start[B] = N.
// ---------------------------------------------------------------------------
__global__ void seg_bounds(const int* __restrict__ batch, int* __restrict__ seg_start,
                           int N, int B) {
    int g = blockIdx.x * blockDim.x + threadIdx.x;
    if (g > B) return;
    int lo = 0, hi = N;
    while (lo < hi) {
        int mid = (lo + hi) >> 1;
        if (batch[mid] < g) lo = mid + 1; else hi = mid;
    }
    seg_start[g] = lo;
}

// ---------------------------------------------------------------------------
// K3: per-graph softmax + weighted mean pool. One block (256 threads) per graph;
// thread t owns feature dim t. exp() values staged in LDS per 256-row chunk.
// ---------------------------------------------------------------------------
__global__ __launch_bounds__(256) void pool_kernel(
    const float* __restrict__ x, const float* __restrict__ scores,
    const int* __restrict__ seg_start, float* __restrict__ out)
{
    __shared__ float red[256];
    __shared__ float ebuf[256];
    const int g = blockIdx.x;
    const int t = threadIdx.x;
    const int s = seg_start[g];
    const int e = seg_start[g + 1];
    const int cnt = e - s;
    if (cnt <= 0) { out[(size_t)g * D_DIM + t] = 0.f; return; }

    // pass 1: segment max
    float m = -3.402823466e38f;
    for (int i = s + t; i < e; i += 256) m = fmaxf(m, scores[i]);
    red[t] = m; __syncthreads();
    for (int off = 128; off > 0; off >>= 1) {
        if (t < off) red[t] = fmaxf(red[t], red[t + off]);
        __syncthreads();
    }
    m = red[0]; __syncthreads();

    // pass 2: sum of exp
    float se = 0.f;
    for (int i = s + t; i < e; i += 256) se += fast_exp(scores[i] - m);
    red[t] = se; __syncthreads();
    for (int off = 128; off > 0; off >>= 1) {
        if (t < off) red[t] += red[t + off];
        __syncthreads();
    }
    const float denom = red[0]; __syncthreads();

    // pass 3: out[g][t] = (1/cnt) * sum_i (e_i/denom) * x[i][t]
    float accv = 0.f;
    for (int c0 = s; c0 < e; c0 += 256) {
        const int len = min(256, e - c0);
        if (t < len) ebuf[t] = fast_exp(scores[c0 + t] - m);
        __syncthreads();
        for (int j = 0; j < len; ++j)
            accv += ebuf[j] * x[(size_t)(c0 + j) * D_DIM + t];
        __syncthreads();
    }
    out[(size_t)g * D_DIM + t] = accv / denom / (float)cnt;
}

// ---------------------------------------------------------------------------
extern "C" void kernel_launch(void* const* d_in, const int* in_sizes, int n_in,
                              void* d_out, int out_size, void* d_ws, size_t ws_size,
                              hipStream_t stream) {
    const float* x     = (const float*)d_in[0];
    const int*   batch = (const int*)d_in[1];   // harness delivers integer inputs as int32
    const float* w1    = (const float*)d_in[3];
    const float* b1    = (const float*)d_in[4];
    const float* w2    = (const float*)d_in[5];
    const float* b2    = (const float*)d_in[6];
    float*       out   = (float*)d_out;

    const int N = in_sizes[1];            // 400000
    const int B = out_size / D_DIM;       // 1024

    // workspace layout: [wb 64KB][scores N*4B][seg_start (B+1)*4B]
    char*   ws     = (char*)d_ws;
    __bf16* wb     = (__bf16*)ws;
    float*  scores = (float*)(ws + 65536);
    size_t  segOff = 65536 + (((size_t)N * 4 + 15) & ~(size_t)15);
    int*    seg    = (int*)(ws + segOff);

    swizzle_w1<<<(KT * NT * 32 * 16) / 256, 256, 0, stream>>>(w1, wb);

    const int nblk = (N + ROWS_PER_BLOCK - 1) / ROWS_PER_BLOCK;
    score_kernel<<<nblk, 256, 0, stream>>>(x, wb, b1, w2, b2, scores, N);

    seg_bounds<<<(B + 1 + 255) / 256, 256, 0, stream>>>(batch, seg, N, B);

    pool_kernel<<<B, 256, 0, stream>>>(x, scores, seg, out);
}
